// EncodingLayer_63436666962232
// MI455X (gfx1250) — matrix operand order repaired and checked by reading, not compiled, not run on hardware
//
#include <hip/hip_runtime.h>
#include <hip/hip_bf16.h>

typedef __attribute__((ext_vector_type(16))) _Float16 v16h;
typedef __attribute__((ext_vector_type(8)))  _Float16 v8h;
typedef __attribute__((ext_vector_type(8)))  float    v8f;
typedef __attribute__((ext_vector_type(4)))  unsigned int v4u;
typedef __attribute__((ext_vector_type(8)))  int v8i;
typedef __attribute__((ext_vector_type(4)))  int v4i;

#define NB    2
#define NS    2048
#define NHID  128
#define NHEAD 8
#define BS    (NB * NS)       /* 4096 rows */
#define DMODEL (NHEAD * NHID) /* 1024 */
#define KMAX  1024            /* largest GEMM K (W_O) -> 32KB LDS B-strip */

// ---------------------------------------------------------------------------
// WMMA fragment helpers (16x16x32 f16, f32 accumulate), wave32 layouts per ISA
// ---------------------------------------------------------------------------
__device__ inline v16h make_v16h(v8h lo, v8h hi) {
  return __builtin_shufflevector(lo, hi, 0,1,2,3,4,5,6,7,8,9,10,11,12,13,14,15);
}

// A matrix 16x32 (MxK), row-major source. lane<16: row=lane, K {0..7,16..23};
// lane>=16: row=lane-16, K {8..15,24..31}.  Two 16B loads per lane.
__device__ inline v16h load_a_frag(const _Float16* A, int lda, int row0, int k0, int lane) {
  int r    = lane & 15;
  int koff = (lane & 16) ? 8 : 0;
  const _Float16* p = A + (size_t)(row0 + r) * lda + k0 + koff;
  v8h lo = *(const v8h*)p;
  v8h hi = *(const v8h*)(p + 16);
  return make_v16h(lo, hi);
}

// B matrix 32x16 (KxN) taken from a PRE-TRANSPOSED source Bt[N][K] (row-major,
// leading dim ldb over K).  lane = column n (mod 16); lanes 0-15 hold K {0..7,16..23},
// lanes 16-31 hold K {8..15,24..31} (matches A fragment).  Two 16B loads per lane.
__device__ inline v16h load_b_frag(const _Float16* Bt, int ldb, int col0, int k0, int lane) {
  int n    = lane & 15;
  int koff = (lane & 16) ? 8 : 0;
  const _Float16* p = Bt + (size_t)(col0 + n) * ldb + k0 + koff;
  v8h lo = *(const v8h*)p;
  v8h hi = *(const v8h*)(p + 16);
  return make_v16h(lo, hi);
}

#define WMMA(a, b, c) \
  __builtin_amdgcn_wmma_f32_16x16x32_f16(false, (a), false, (b), (short)0, (c), false, false)

// ---------------------------------------------------------------------------
// Prep: x -> f16, all weights -> transposed f16 so B-fragments are contiguous
// ---------------------------------------------------------------------------
__global__ void prep_kernel(const float* __restrict__ x,
                            const float* __restrict__ Wq, const float* __restrict__ Wk,
                            const float* __restrict__ Wv, const float* __restrict__ Wo,
                            const float* __restrict__ W1, const float* __restrict__ W2,
                            _Float16* __restrict__ xh,
                            _Float16* __restrict__ WqT, _Float16* __restrict__ WkT,
                            _Float16* __restrict__ WvT, _Float16* __restrict__ WoT,
                            _Float16* __restrict__ W1T, _Float16* __restrict__ W2T) {
  const int NXH = BS * NHID;            // 524288
  const int NWH = NHEAD * NHID * NHID;  // 131072
  const int NWO = DMODEL * NHID;        // 131072
  const int NW1 = NHID * 2 * NHID;      // 32768
  const int total = NXH + 3 * NWH + NWO + 2 * NW1;
  for (int i = blockIdx.x * blockDim.x + threadIdx.x; i < total;
       i += gridDim.x * blockDim.x) {
    int j = i;
    if (j < NXH) { xh[j] = (_Float16)x[j]; continue; }
    j -= NXH;
    if (j < NWH) { int h=j>>14, r=j&16383, e=r>>7, d=r&127;
                   WqT[j] = (_Float16)Wq[(h<<14)+(d<<7)+e]; continue; }
    j -= NWH;
    if (j < NWH) { int h=j>>14, r=j&16383, e=r>>7, d=r&127;
                   WkT[j] = (_Float16)Wk[(h<<14)+(d<<7)+e]; continue; }
    j -= NWH;
    if (j < NWH) { int h=j>>14, r=j&16383, e=r>>7, d=r&127;
                   WvT[j] = (_Float16)Wv[(h<<14)+(d<<7)+e]; continue; }
    j -= NWH;
    if (j < NWO) { int n=j>>10, k=j&1023; WoT[j] = (_Float16)Wo[k*NHID+n]; continue; }
    j -= NWO;
    if (j < NW1) { int n=j>>7, k=j&127;  W1T[j] = (_Float16)W1[k*256+n]; continue; }
    j -= NW1;
    { int n=j>>8, k=j&255; W2T[j] = (_Float16)W2[k*NHID+n]; }
  }
}

// ---------------------------------------------------------------------------
// WMMA GEMM v2:
//   - block = 8 waves, all sharing one 16-column strip of Bt (requires ldb==K)
//   - B strip (16 x K halves, contiguous) staged into LDS once per block via
//     the Tensor Data Mover (tensor_load_to_lds + s_wait_tensorcnt), with a
//     cooperative-copy fallback if the builtin is unavailable
//   - each wave register-blocks 4 M-tiles (64x16 output, 4 accumulators),
//     so each B fragment (from LDS) feeds 4 WMMAs
// Fused epilogue: +bias, relu, +resid; stores f32 / f16 / transposed-f16.
// ---------------------------------------------------------------------------
__global__ void gemm16_kernel(const _Float16* __restrict__ A, int lda,
                              const _Float16* __restrict__ Bt,
                              const float* __restrict__ bias,
                              float* __restrict__ Cf, _Float16* __restrict__ Ch, int ldc,
                              _Float16* __restrict__ Ct, int ldt,
                              const float* __restrict__ resid,
                              int Mblocks, int Kd, int relu) {
  __shared__ __align__(32) _Float16 sB[16 * KMAX];   // 32 KB max strip

  int lane = threadIdx.x & 31;
  int wv   = threadIdx.x >> 5;
  int mb   = blockIdx.x % Mblocks;
  int nt   = blockIdx.x / Mblocks;
  int row0 = (mb * 8 + wv) * 64;       // 64 rows per wave
  int col0 = nt * 16;

  // ---- stage B strip: Bt[col0..col0+15][0..Kd) is 16*Kd contiguous halves --
#if __has_builtin(__builtin_amdgcn_tensor_load_to_lds)
  if (threadIdx.x < 32) {              // one wave drives the TDM
    unsigned ldsoff = (unsigned)(size_t)(&sB[0]);               // LDS byte offset
    unsigned long long ga =
        (unsigned long long)(size_t)(Bt + (size_t)col0 * Kd);   // tile start
    // D# group 0: count=1 | lds_addr | global_addr[56:0] | type=2
    v4u g0;
    g0.x = 1u;
    g0.y = ldsoff;
    g0.z = (unsigned)(ga & 0xffffffffu);
    g0.w = (unsigned)((ga >> 32) & 0x01ffffffu) | (2u << 30);
    // D# group 1: data_size=2B; tensor_dim0=Kd; tensor_dim1=16;
    //             tile_dim0=Kd; tile_dim1=16; dim0_stride=Kd
    v8i g1;
    g1[0] = (int)(1u << 16);                                    // data_size code 1
    g1[1] = (int)(((unsigned)Kd & 0xffffu) << 16);              // tensor_dim0 lo
    g1[2] = (int)((((unsigned)Kd >> 16) & 0xffffu) | (16u << 16)); // dim0 hi | dim1 lo
    g1[3] = (int)(((unsigned)Kd & 0xffffu) << 16);              // dim1 hi=0 | tile_dim0
    g1[4] = 16;                                                 // tile_dim1=16, tile_dim2=0
    g1[5] = Kd;                                                 // dim0_stride lo32
    g1[6] = 0;                                                  // dim0_stride hi | dim1_stride lo
    g1[7] = 0;
    v4i z4 = {0, 0, 0, 0};                                      // groups 2/3 unused (2D)
    v8i z8 = {0, 0, 0, 0, 0, 0, 0, 0};
    // clang-23 / therock form: 6 args (g0, g1, grp2, grp3, grp4, cpol)
    __builtin_amdgcn_tensor_load_to_lds(g0, g1, z4, z4, z8, 0);
#if __has_builtin(__builtin_amdgcn_s_wait_tensorcnt)
    __builtin_amdgcn_s_wait_tensorcnt(0);
#else
    asm volatile("s_wait_tensorcnt 0x0" ::: "memory");
#endif
  }
#else
  {
    const v8h* src = (const v8h*)(Bt + (size_t)col0 * Kd);
    v8h* dst = (v8h*)&sB[0];
    int n8 = 16 * Kd / 8;
    for (int i = threadIdx.x; i < n8; i += blockDim.x) dst[i] = src[i];
  }
#endif
  __syncthreads();

  const _Float16* sBp = &sB[0];
  v8f acc[4] = {};
  for (int k0 = 0; k0 < Kd; k0 += 32) {
    v16h bf = load_b_frag(sBp, Kd, 0, k0, lane);   // from LDS, shared by block
#pragma unroll
    for (int r4 = 0; r4 < 4; r4++)
      acc[r4] = WMMA(load_a_frag(A, lda, row0 + r4 * 16, k0, lane), bf, acc[r4]);
  }

  int mrow = (lane & 16) ? 8 : 0;      // D layout: VGPR r -> row r(+8), lane -> col
  int col  = col0 + (lane & 15);
  float bv = bias ? bias[col] : 0.f;
#pragma unroll
  for (int r4 = 0; r4 < 4; r4++) {
#pragma unroll
    for (int r = 0; r < 8; r++) {
      int row = row0 + r4 * 16 + r + mrow;
      float v = acc[r4][r] + bv;
      if (relu)  v = fmaxf(v, 0.f);
      if (resid) v += resid[(size_t)row * ldc + col];
      if (Cf) Cf[(size_t)row * ldc + col] = v;
      if (Ch) Ch[(size_t)row * ldc + col] = (_Float16)v;
      if (Ct) Ct[(size_t)col * ldt + row] = (_Float16)v;   // used to build V^T
    }
  }
}

// ---------------------------------------------------------------------------
// Flash attention: 1 wave per 16-query tile, online softmax, O held in regs.
// Q,K stored [h][b*S+t][e] (row-major, ld=128); V stored transposed
// Vt[h][e][b*S+t] (ld=BS) so P*V B-fragments are contiguous per lane.
// ---------------------------------------------------------------------------
#define ATT_NW 4
__global__ void attn_kernel(const _Float16* __restrict__ Q,
                            const _Float16* __restrict__ K,
                            const _Float16* __restrict__ Vt,
                            _Float16* __restrict__ O) {
  __shared__ __align__(32) float    sS[ATT_NW][16][32];
  __shared__ __align__(32) _Float16 sP[ATT_NW][16][32];
  __shared__ float sC[ATT_NW][16];
  __shared__ float sL[ATT_NW][16];

  int lane = threadIdx.x & 31;
  int w    = threadIdx.x >> 5;
  int tile = blockIdx.x * ATT_NW + w;         // 0 .. NHEAD*NB*(NS/16)-1
  int h    = tile / (NB * (NS / 16));
  int rem  = tile % (NB * (NS / 16));
  int b    = rem / (NS / 16);
  int qt   = rem % (NS / 16);

  const _Float16* Qb = Q  + ((size_t)(h * NB + b) * NS) * NHID;
  const _Float16* Kb = K  + ((size_t)(h * NB + b) * NS) * NHID;
  const _Float16* Vb = Vt + (size_t)h * NHID * BS + (size_t)b * NS;
  int row0 = qt * 16;
  int mrow = (lane & 16) ? 8 : 0;
  int ncol = lane & 15;

  v16h qf[4];
#pragma unroll
  for (int kk = 0; kk < 4; kk++) qf[kk] = load_a_frag(Qb, NHID, row0, kk * 32, lane);

  v8f acc[8] = {};
  float m = -1e30f, l = 0.f;
  const float scale = 0.08838834764831845f;   // 1/sqrt(128)

  for (int t0 = 0; t0 < NS; t0 += 32) {
    // scores for 32 keys: two 16x16 tiles, K-dim = 128 (4 WMMA steps each)
    v8f c0 = {}, c1 = {};
#pragma unroll
    for (int kk = 0; kk < 4; kk++) {
      c0 = WMMA(qf[kk], load_b_frag(Kb, NHID, t0,      kk * 32, lane), c0);
      c1 = WMMA(qf[kk], load_b_frag(Kb, NHID, t0 + 16, kk * 32, lane), c1);
    }
#pragma unroll
    for (int r = 0; r < 8; r++) {
      sS[w][r + mrow][ncol]      = c0[r] * scale;
      sS[w][r + mrow][16 + ncol] = c1[r] * scale;
    }
    __syncthreads();

    // online softmax stats: lane = query row
    if (lane < 16) {
      float tm = -1e30f;
      for (int j = 0; j < 32; j++) tm = fmaxf(tm, sS[w][lane][j]);
      float mn   = fmaxf(m, tm);
      float corr = __expf(m - mn);
      float ts   = 0.f;
      for (int j = 0; j < 32; j++) {
        float p = __expf(sS[w][lane][j] - mn);
        sP[w][lane][j] = (_Float16)p;
        ts += p;
      }
      l = l * corr + ts;
      m = mn;
      sC[w][lane] = corr;
    }
    __syncthreads();

    // rescale running O accumulators
    float cf[8];
#pragma unroll
    for (int r = 0; r < 8; r++) cf[r] = sC[w][r + mrow];
#pragma unroll
    for (int nt = 0; nt < 8; nt++)
#pragma unroll
      for (int r = 0; r < 8; r++) acc[nt][r] *= cf[r];

    // O += P(16x32) * V(32x128): LDS bounce doubles as D->A layout transpose
    v16h pf = load_a_frag(&sP[w][0][0], 32, 0, 0, lane);
#pragma unroll
    for (int nt = 0; nt < 8; nt++)
      acc[nt] = WMMA(pf, load_b_frag(Vb, BS, nt * 16, t0, lane), acc[nt]);
    __syncthreads();
  }

  if (lane < 16) sL[w][lane] = l;
  __syncthreads();
#pragma unroll
  for (int nt = 0; nt < 8; nt++) {
#pragma unroll
    for (int r = 0; r < 8; r++) {
      float inv  = 1.f / sL[w][r + mrow];
      size_t row = (size_t)b * NS + row0 + r + mrow;
      O[row * DMODEL + h * NHID + nt * 16 + ncol] = (_Float16)(acc[nt][r] * inv);
    }
  }
}

// ---------------------------------------------------------------------------
// LayerNorm over last dim (128): one wave32 per row, shuffle reductions.
// ---------------------------------------------------------------------------
__device__ inline float wave_sum(float v) {
#pragma unroll
  for (int off = 16; off >= 1; off >>= 1) v += __shfl_xor(v, off, 32);
  return v;
}

__global__ void ln_kernel(const float* __restrict__ X,
                          const float* __restrict__ g, const float* __restrict__ be,
                          float* __restrict__ Yf, _Float16* __restrict__ Yh, int rows) {
  int row = blockIdx.x * 8 + (threadIdx.x >> 5);
  if (row >= rows) return;
  int lane = threadIdx.x & 31;
  const float* xp = X + (size_t)row * NHID;
  float v[4];
#pragma unroll
  for (int i = 0; i < 4; i++) v[i] = xp[lane + 32 * i];
  float mean = wave_sum(v[0] + v[1] + v[2] + v[3]) * (1.f / NHID);
  float sq = 0.f;
#pragma unroll
  for (int i = 0; i < 4; i++) { float d = v[i] - mean; sq += d * d; }
  float var = wave_sum(sq) * (1.f / NHID);
  float inv = rsqrtf(var + 1e-5f);
#pragma unroll
  for (int i = 0; i < 4; i++) {
    int c = lane + 32 * i;
    float o = (v[i] - mean) * inv * g[c] + be[c];
    if (Yf) Yf[(size_t)row * NHID + c] = o;
    if (Yh) Yh[(size_t)row * NHID + c] = (_Float16)o;
  }
}

// ---------------------------------------------------------------------------
// Host launch
// ---------------------------------------------------------------------------
extern "C" void kernel_launch(void* const* d_in, const int* in_sizes, int n_in,
                              void* d_out, int out_size, void* d_ws, size_t ws_size,
                              hipStream_t stream) {
  (void)in_sizes; (void)n_in; (void)out_size; (void)ws_size;
  const float* x   = (const float*)d_in[0];
  const float* Wq  = (const float*)d_in[1];
  const float* bq  = (const float*)d_in[2];
  const float* Wk  = (const float*)d_in[3];
  const float* bk  = (const float*)d_in[4];
  const float* Wv  = (const float*)d_in[5];
  const float* bv  = (const float*)d_in[6];
  const float* Wo  = (const float*)d_in[7];
  const float* bo  = (const float*)d_in[8];
  const float* W1  = (const float*)d_in[9];
  const float* b1  = (const float*)d_in[10];
  const float* W2  = (const float*)d_in[11];
  const float* b2  = (const float*)d_in[12];
  const float* g1  = (const float*)d_in[13];
  const float* be1 = (const float*)d_in[14];
  const float* g2  = (const float*)d_in[15];
  const float* be2 = (const float*)d_in[16];
  float* out = (float*)d_out;

  char* w = (char*)d_ws;
  size_t off = 0;
  auto alloc = [&](size_t bytes) -> char* {
    char* p = w + off;
    off = (off + bytes + 255) & ~(size_t)255;
    return p;
  };
  _Float16* xh   = (_Float16*)alloc((size_t)BS * NHID * 2);
  _Float16* WqT  = (_Float16*)alloc((size_t)NHEAD * NHID * NHID * 2);
  _Float16* WkT  = (_Float16*)alloc((size_t)NHEAD * NHID * NHID * 2);
  _Float16* WvT  = (_Float16*)alloc((size_t)NHEAD * NHID * NHID * 2);
  _Float16* WoT  = (_Float16*)alloc((size_t)DMODEL * NHID * 2);
  _Float16* W1T  = (_Float16*)alloc((size_t)NHID * 256 * 2);
  _Float16* W2T  = (_Float16*)alloc((size_t)256 * NHID * 2);
  _Float16* Qb   = (_Float16*)alloc((size_t)NHEAD * BS * NHID * 2);
  _Float16* Kb   = (_Float16*)alloc((size_t)NHEAD * BS * NHID * 2);
  _Float16* Vt   = (_Float16*)alloc((size_t)NHEAD * NHID * BS * 2);
  _Float16* obuf = (_Float16*)alloc((size_t)BS * DMODEL * 2);
  float*    ybuf = (float*)   alloc((size_t)BS * NHID * 4);
  float*    yln  = (float*)   alloc((size_t)BS * NHID * 4);
  _Float16* ylnh = (_Float16*)alloc((size_t)BS * NHID * 2);
  _Float16* h1b  = (_Float16*)alloc((size_t)BS * 256 * 2);
  float*    zbuf = (float*)   alloc((size_t)BS * NHID * 4);

  prep_kernel<<<1024, 256, 0, stream>>>(x, Wq, Wk, Wv, Wo, W1, W2,
                                        xh, WqT, WkT, WvT, WoT, W1T, W2T);

  // M is 4096 for every GEMM; 8 waves x 64 rows = 512 rows per block.
  auto gemm = [&](const _Float16* A, int lda, const _Float16* Bt,
                  const float* bias, float* Cf, _Float16* Ch, int ldc,
                  _Float16* Ct, int ldt, const float* resid,
                  int Ntiles, int Kd, int relu) {
    int Mblocks = BS / 512;                       // 8
    gemm16_kernel<<<Mblocks * Ntiles, 256, 0, stream>>>(
        A, lda, Bt, bias, Cf, Ch, ldc, Ct, ldt, resid, Mblocks, Kd, relu);
  };

  // QKV projections per head (V stored transposed for the P*V stage)
  for (int h = 0; h < NHEAD; h++) {
    gemm(xh, NHID, WqT + (size_t)h * NHID * NHID, bq + h * NHID,
         nullptr, Qb + (size_t)h * BS * NHID, NHID, nullptr, 0, nullptr,
         NHID / 16, NHID, 0);
    gemm(xh, NHID, WkT + (size_t)h * NHID * NHID, bk + h * NHID,
         nullptr, Kb + (size_t)h * BS * NHID, NHID, nullptr, 0, nullptr,
         NHID / 16, NHID, 0);
    gemm(xh, NHID, WvT + (size_t)h * NHID * NHID, bv + h * NHID,
         nullptr, nullptr, 0, Vt + (size_t)h * NHID * BS, BS, nullptr,
         NHID / 16, NHID, 0);
  }

  // Flash attention
  {
    int tiles = NHEAD * NB * (NS / 16);   // 2048
    attn_kernel<<<tiles / ATT_NW, ATT_NW * 32, 0, stream>>>(Qb, Kb, Vt, obuf);
  }

  // W_O projection -> y (f32)
  gemm(obuf, DMODEL, WoT, bo, ybuf, nullptr, NHID, nullptr, 0, nullptr,
       NHID / 16, DMODEL, 0);

  // LN1 -> yln (f32, residual source) + ylnh (f16, FFN input)
  ln_kernel<<<BS / 8, 256, 0, stream>>>(ybuf, g1, be1, yln, ylnh, BS);

  // FFN: relu(y@W1+b1) -> h1 (f16)
  gemm(ylnh, NHID, W1T, b1, nullptr, h1b, 256, nullptr, 0, nullptr,
       256 / 16, NHID, 1);
  // z = yln + relu(h1@W2+b2)  (f32)
  gemm(h1b, 256, W2T, b2, zbuf, nullptr, NHID, nullptr, 0, yln,
       NHID / 16, 256, 1);

  // LN2 -> final output
  ln_kernel<<<BS / 8, 256, 0, stream>>>(zbuf, g2, be2, out, nullptr, BS);
}
